// MoleculeDenoiser_56023553409118
// MI455X (gfx1250) — matrix operand order repaired and checked by reading, not compiled, not run on hardware
//
#include <hip/hip_runtime.h>
#include <math.h>

// ---------------------------------------------------------------------------
// Types for WMMA (gfx1250, wave32)
// ---------------------------------------------------------------------------
typedef __attribute__((ext_vector_type(16))) _Float16 v16h;
typedef __attribute__((ext_vector_type(8)))  float    v8f;

union FragU { unsigned u[8]; v16h v; };

static constexpr int NN   = 8192;    // atoms
static constexpr int EBN  = 16128;   // bond edges
static constexpr int ESN  = 81920;   // knn edges
static constexpr int KNB  = 10;      // knn K

// ---------------------------------------------------------------------------
// WMMA GEMM:  Out[M,N] = act( gatherA[M,K] @ Wt[K,N] + bias )   (f16 in, f32 acc)
// A operand is a concatenation of up to 3 column-segments, each with its own
// base pointer, optional row-index array (gather) and leading dimension.
// Weight tiles are staged via gfx1250 async DMA (global_load_async_to_lds)
// whenever the tile is fully in-bounds (K%32==0, N==128).
// ---------------------------------------------------------------------------
template<int NTV>
__global__ __launch_bounds__(128)
void gemm_wmma_kernel(const float* A0, const int* I0, int w0, int lda0,
                      const float* A1, const int* I1, int w1, int lda1,
                      const float* A2, const int* I2, int w2, int lda2,
                      const float* Wt, const float* bias,
                      float* Out, int ldo,
                      int M, int N, int K, int act, int accum)
{
    __shared__ _Float16 As[64][34];      // [m][k] tile, fp16
    __shared__ _Float16 Bs[128][34];     // [n][k] tile (transposed), fp16
    __shared__ float    Bstage[32 * 128];// fp32 async staging for weight tile

    const int tid  = threadIdx.x;
    const int lane = tid & 31;
    const int wid  = tid >> 5;
    const int m0   = blockIdx.x * 64;
    const int hf   = lane >> 4;          // half-wave (0/1)
    const int l16  = lane & 15;

    const bool asyncB = ((K & 31) == 0) && (N == 128);  // block-uniform

    v8f acc[NTV];
#pragma unroll
    for (int i = 0; i < NTV; ++i)
#pragma unroll
        for (int e = 0; e < 8; ++e) acc[i][e] = 0.0f;

    for (int k0 = 0; k0 < K; k0 += 32) {
        // ---- stage A tile (gathered, segmented, fp32 -> fp16) ----
        for (int idx = tid; idx < 64 * 32; idx += 128) {
            int r = idx >> 5, kk = idx & 31;
            int gm = m0 + r, gk = k0 + kk;
            float v = 0.0f;
            if (gm < M && gk < K) {
                if (gk < w0) {
                    int row = I0 ? I0[gm] : gm;
                    v = A0[(size_t)row * lda0 + gk];
                } else if (gk < w0 + w1) {
                    int row = I1 ? I1[gm] : gm;
                    v = A1[(size_t)row * lda1 + (gk - w0)];
                } else {
                    int row = I2 ? I2[gm] : gm;
                    v = A2[(size_t)row * lda2 + (gk - w0 - w1)];
                }
            }
            As[r][kk] = (_Float16)v;
        }

        if (asyncB) {
            // ---- async DMA the full 32x128 fp32 weight tile into LDS ----
            // Each thread moves 32 floats = 128 B = 8 x b128 chunks.
            const unsigned base = (unsigned)tid * 32u;
            unsigned           la = (unsigned)(size_t)&Bstage[base];       // LDS offset = addr[31:0]
            unsigned long long ga = (unsigned long long)(size_t)(Wt + (size_t)k0 * 128 + base);
#pragma unroll
            for (int c = 0; c < 8; ++c) {
                asm volatile("global_load_async_to_lds_b128 %0, %1, off"
                             :: "v"(la), "v"(ga) : "memory");
                la += 16u; ga += 16ull;
            }
#if __has_builtin(__builtin_amdgcn_s_wait_asynccnt)
            __builtin_amdgcn_s_wait_asynccnt(0);
#else
            asm volatile("s_wait_asynccnt 0x0" ::: "memory");
#endif
            __syncthreads();   // all waves' async stores to LDS complete
            // convert staged fp32 -> fp16, transposed into Bs[n][k]
            for (int idx = tid; idx < 32 * 128; idx += 128) {
                int kk = idx >> 7, n = idx & 127;
                Bs[n][kk] = (_Float16)Bstage[idx];
            }
        } else {
            // ---- guarded direct path (K tails / small N) ----
            for (int idx = tid; idx < 128 * 32; idx += 128) {
                int n = idx >> 5, kk = idx & 31;
                int gk = k0 + kk;
                float v = 0.0f;
                if (n < N && gk < K) v = Wt[(size_t)gk * N + n];
                Bs[n][kk] = (_Float16)v;
            }
            if (k0 + 32 < K && tid == 0)
                __builtin_prefetch(&Wt[(size_t)(k0 + 32) * N], 0, 1);
        }
        __syncthreads();

        // ---- A fragment: ISA 16-bit A 16x32 layout ----
        // lane L (row = L%16, half = L/16): VGPR j holds K = (j<4 ? 2j : 2j+8)+half*8, +1
        FragU a;
#pragma unroll
        for (int j = 0; j < 8; ++j) {
            int kk = ((j < 4) ? (2 * j) : (2 * j + 8)) + hf * 8;
            a.u[j] = *(const unsigned*)&As[(wid << 4) + l16][kk];
        }
        // ---- B fragments + WMMA ----
#pragma unroll
        for (int nt = 0; nt < NTV; ++nt) {
            FragU b;
#pragma unroll
            for (int j = 0; j < 8; ++j) {
                // B 32x16: VGPR j holds K = (j<4 ? 2j : 2j+8)+half*8, +1; N = lane%16
                b.u[j] = *(const unsigned*)&Bs[(nt << 4) + l16][((j < 4) ? (2 * j) : (2 * j + 8)) + hf * 8];
            }
            acc[nt] = __builtin_amdgcn_wmma_f32_16x16x32_f16(
                false, a.v, false, b.v, (short)0, acc[nt], false, false);
        }
        __syncthreads();
    }

    // ---- epilogue: C/D layout: VGPR r -> M = r + 8*half, N = lane%16 ----
#pragma unroll
    for (int nt = 0; nt < NTV; ++nt) {
        int gn = (nt << 4) + l16;
#pragma unroll
        for (int r = 0; r < 8; ++r) {
            int gm = m0 + (wid << 4) + r + (hf << 3);
            if (gm < M && gn < N) {
                float v = acc[nt][r];
                if (bias) v += bias[gn];
                if (act == 1)      v = v > 0.0f ? v : 0.0f;
                else if (act == 2) v = 1.0f / (1.0f + expf(-v));
                float* o = Out + (size_t)gm * ldo + gn;
                if (accum) *o += v; else *o = v;
            }
        }
    }
}

// ---------------------------------------------------------------------------
// Utility kernels
// ---------------------------------------------------------------------------
__global__ void zero_f32(float* p, int n) {
    int t = blockIdx.x * blockDim.x + threadIdx.x;
    if (t < n) p[t] = 0.0f;
}
__global__ void copy_f32(float* d, const float* s, int n) {
    int t = blockIdx.x * blockDim.x + threadIdx.x;
    if (t < n) d[t] = s[t];
}
__global__ void iota_div(int* p, int n, int dv) {
    int t = blockIdx.x * blockDim.x + threadIdx.x;
    if (t < n) p[t] = t / dv;
}

// LayerNorm over 128 channels (optional residual with ld=128)
__global__ __launch_bounds__(128)
void ln128_kernel(float* io, const float* resid, const float* g, const float* b, int ld)
{
    __shared__ float red[128];
    int row = blockIdx.x;
    int c   = threadIdx.x;
    float* p = io + (size_t)row * ld;
    float v  = p[c];
    if (resid) v += resid[(size_t)row * 128 + c];
    red[c] = v; __syncthreads();
    for (int s = 64; s > 0; s >>= 1) { if (c < s) red[c] += red[c + s]; __syncthreads(); }
    float mean = red[0] * (1.0f / 128.0f);
    __syncthreads();
    float d = v - mean;
    red[c] = d * d; __syncthreads();
    for (int s = 64; s > 0; s >>= 1) { if (c < s) red[c] += red[c + s]; __syncthreads(); }
    float var = red[0] * (1.0f / 128.0f);
    p[c] = d * rsqrtf(var + 1e-5f) * g[c] + b[c];
}

// bond rbf: linspace(0,3,32), sig = 3/32; len = ||p[s]-p[d] + 1e-8||
__global__ void bond_rbf_kernel(const float* pos, const int* src, const int* dst,
                                float* rbf, int E)
{
    int e = blockIdx.x * blockDim.x + threadIdx.x;
    if (e >= E) return;
    int s = src[e], d = dst[e];
    float dx = pos[s * 3 + 0] - pos[d * 3 + 0] + 1e-8f;
    float dy = pos[s * 3 + 1] - pos[d * 3 + 1] + 1e-8f;
    float dz = pos[s * 3 + 2] - pos[d * 3 + 2] + 1e-8f;
    float len = sqrtf(dx * dx + dy * dy + dz * dz);
    const float inv_sig = 32.0f / 3.0f;
    for (int r = 0; r < 32; ++r) {
        float mu = 3.0f * (float)r / 31.0f;
        float t  = (len - mu) * inv_sig;
        rbf[(size_t)e * 32 + r] = expf(-t * t);
    }
}

// KNN (brute force per atom, matches top_k(-D) tie-break = lowest index),
// fused with slen + rbf linspace(0,10,32), sig = 10/32.
__global__ void knn_kernel(const float* pos, const int* batch, int* ssrc,
                           float* rbf, int n)
{
    int i = blockIdx.x * blockDim.x + threadIdx.x;
    if (i >= n) return;
    float px = pos[i * 3 + 0], py = pos[i * 3 + 1], pz = pos[i * 3 + 2];
    float sqi = px * px + py * py + pz * pz;
    int bi = batch[i];
    float bd[KNB]; int bidx[KNB];
    for (int t = 0; t < KNB; ++t) { bd[t] = 1e30f; bidx[t] = 0; }
    for (int j = 0; j < n; ++j) {
        if (j == i || batch[j] != bi) continue;
        float qx = pos[j * 3 + 0], qy = pos[j * 3 + 1], qz = pos[j * 3 + 2];
        float D = sqi + (qx * qx + qy * qy + qz * qz)
                - 2.0f * (px * qx + py * qy + pz * qz);
        if (D < bd[KNB - 1]) {
            int t = KNB - 1;
            while (t > 0 && D < bd[t - 1]) { bd[t] = bd[t - 1]; bidx[t] = bidx[t - 1]; --t; }
            bd[t] = D; bidx[t] = j;
        }
    }
    const float inv_sig = 32.0f / 10.0f;
    for (int k = 0; k < KNB; ++k) {
        int j = bidx[k];
        int e = i * KNB + k;
        ssrc[e] = j;
        float dx = pos[j * 3 + 0] - px + 1e-8f;
        float dy = pos[j * 3 + 1] - py + 1e-8f;
        float dz = pos[j * 3 + 2] - pz + 1e-8f;
        float len = sqrtf(dx * dx + dy * dy + dz * dz);
        for (int r = 0; r < 32; ++r) {
            float mu = 10.0f * (float)r / 31.0f;
            float t  = (len - mu) * inv_sig;
            rbf[(size_t)e * 32 + r] = expf(-t * t);
        }
    }
}

// ---- segment softmax over dst (H=4 heads), ordered-int atomic max ----
__device__ __forceinline__ int   fenc(float f) { int i = __float_as_int(f); return i >= 0 ? i : (i ^ 0x7fffffff); }
__device__ __forceinline__ float fdec(int i)   { return __int_as_float(i >= 0 ? i : (i ^ 0x7fffffff)); }

__global__ void seg_init_kernel(int* nmax, float* nden, int n) {
    int t = blockIdx.x * blockDim.x + threadIdx.x;
    if (t < n) { nmax[t] = (int)0x80000000; nden[t] = 0.0f; }
}
__global__ void seg_max_kernel(const float* logits, const int* dst, int* nmax, int E) {
    int t = blockIdx.x * blockDim.x + threadIdx.x;
    if (t >= E * 4) return;
    int e = t >> 2, h = t & 3;
    atomicMax(&nmax[dst[e] * 4 + h], fenc(logits[t]));
}
__global__ void seg_exp_kernel(float* logits, const int* dst, const int* nmax, float* nden, int E) {
    int t = blockIdx.x * blockDim.x + threadIdx.x;
    if (t >= E * 4) return;
    int e = t >> 2, h = t & 3;
    float m = fdec(nmax[dst[e] * 4 + h]);
    float v = expf(logits[t] - m);
    logits[t] = v;
    atomicAdd(&nden[dst[e] * 4 + h], v);
}
__global__ void seg_norm_kernel(float* logits, const int* dst, const float* nden, int E) {
    int t = blockIdx.x * blockDim.x + threadIdx.x;
    if (t >= E * 4) return;
    int e = t >> 2, h = t & 3;
    logits[t] = logits[t] / (nden[dst[e] * 4 + h] + 1e-9f);
}

// msum[dst] += alpha[e,h] * vnode[src]   (rotation cancels: P^T P = I)
__global__ void scatter_msg_kernel(const float* alpha, const float* vnode,
                                   const int* src, const int* dst,
                                   float* msum, int E)
{
    int t = blockIdx.x * blockDim.x + threadIdx.x;
    if (t >= E * 512) return;
    int e = t >> 9, i = t & 511;
    int h = (i & 127) >> 5;
    float a = alpha[e * 4 + h];
    atomicAdd(&msum[(size_t)dst[e] * 512 + i], a * vnode[(size_t)src[e] * 512 + i]);
}

// x[:,0] += f0;  x[:,1:4] *= (1 + sigmoid_gate)
__global__ void gate_apply_kernel(float* x, const float* f0, const float* gate, int n)
{
    int t = blockIdx.x * blockDim.x + threadIdx.x;
    if (t >= n * 128) return;
    int node = t >> 7, c = t & 127;
    size_t base = (size_t)node * 512 + c;
    x[base] += f0[t];
    float g = 1.0f + gate[t];
    x[base + 128] *= g;
    x[base + 256] *= g;
    x[base + 384] *= g;
}

// pos[n,d] += sum_c x[n,1+d,c] * g[n,c]
__global__ __launch_bounds__(32)
void pos_update_kernel(float* pos, const float* x, const float* g, int n)
{
    int node = blockIdx.x;
    int lane = threadIdx.x;
    const float* xr = x + (size_t)node * 512;
    const float* gr = g + (size_t)node * 128;
    for (int d = 0; d < 3; ++d) {
        float s = 0.0f;
        for (int c = lane; c < 128; c += 32) s += xr[(d + 1) * 128 + c] * gr[c];
        for (int o = 16; o > 0; o >>= 1) s += __shfl_down(s, o, 32);
        if (lane == 0) pos[node * 3 + d] += s;
    }
}

// ---------------------------------------------------------------------------
// Host-side helpers
// ---------------------------------------------------------------------------
static inline void launch_gemm(hipStream_t st,
    const float* A0, const int* I0, int w0, int lda0,
    const float* A1, const int* I1, int w1, int lda1,
    const float* A2, const int* I2, int w2, int lda2,
    const float* Wt, const float* bias, float* out, int ldo,
    int M, int N, int K, int act, int accum)
{
    dim3 g((M + 63) / 64), b(128);
    if (N <= 16)
        gemm_wmma_kernel<1><<<g, b, 0, st>>>(A0, I0, w0, lda0, A1, I1, w1, lda1,
                                             A2, I2, w2, lda2, Wt, bias, out, ldo,
                                             M, N, K, act, accum);
    else
        gemm_wmma_kernel<8><<<g, b, 0, st>>>(A0, I0, w0, lda0, A1, I1, w1, lda1,
                                             A2, I2, w2, lda2, Wt, bias, out, ldo,
                                             M, N, K, act, accum);
}
static inline void gemm_plain(hipStream_t st, const float* A, int lda,
                              const float* Wt, const float* bias, float* out, int ldo,
                              int M, int N, int K, int act, int accum)
{
    launch_gemm(st, A, nullptr, K, lda, nullptr, nullptr, 0, 0,
                nullptr, nullptr, 0, 0, Wt, bias, out, ldo, M, N, K, act, accum);
}

struct AttnP {
    const float *Wa1, *ba1, *Wa2, *ba2, *Wv, *Wo, *Wf1, *bf1, *Wf2, *bf2, *Wg, *bg;
};

static void run_attn(hipStream_t st, float* x, const float* ef, int E,
                     const int* src, const int* dst, const AttnP& P,
                     float* h1, float* h2, float* logits, int* nmax, float* nden,
                     float* vnode, float* msum)
{
    // logits = relu([x0[src]|x0[dst]|ef] @ Wa1 + ba1) @ Wa2 + ba2
    launch_gemm(st, x, src, 128, 512, x, dst, 128, 512, ef, nullptr, 128, 128,
                P.Wa1, P.ba1, h1, 128, E, 128, 384, 1, 0);
    gemm_plain(st, h1, 128, P.Wa2, P.ba2, logits, 4, E, 4, 128, 0, 0);
    // segment softmax over dst
    seg_init_kernel<<<(NN * 4 + 255) / 256, 256, 0, st>>>(nmax, nden, NN * 4);
    seg_max_kernel <<<(E * 4 + 255) / 256, 256, 0, st>>>(logits, dst, nmax, E);
    seg_exp_kernel <<<(E * 4 + 255) / 256, 256, 0, st>>>(logits, dst, nmax, nden, E);
    seg_norm_kernel<<<(E * 4 + 255) / 256, 256, 0, st>>>(logits, dst, nden, E);
    // vnode = x @ Wv  (per node, rotations cancel)
    gemm_plain(st, x, 128, P.Wv, nullptr, vnode, 128, NN * 4, 128, 128, 0, 0);
    // msum = scatter(alpha * vnode[src])
    zero_f32<<<(NN * 512 + 255) / 256, 256, 0, st>>>(msum, NN * 512);
    scatter_msg_kernel<<<(E * 512 + 255) / 256, 256, 0, st>>>(logits, vnode, src, dst, msum, E);
    // x += msum @ Wo
    gemm_plain(st, msum, 128, P.Wo, nullptr, x, 128, NN * 4, 128, 128, 0, 1);
    // FFN + gate on updated x0
    launch_gemm(st, x, nullptr, 128, 512, nullptr, nullptr, 0, 0, nullptr, nullptr, 0, 0,
                P.Wf1, P.bf1, h1, 128, NN, 128, 128, 1, 0);
    gemm_plain(st, h1, 128, P.Wf2, P.bf2, h2, 128, NN, 128, 128, 0, 0);
    launch_gemm(st, x, nullptr, 128, 512, nullptr, nullptr, 0, 0, nullptr, nullptr, 0, 0,
                P.Wg, P.bg, msum, 128, NN, 128, 128, 2, 0);  // sigmoid
    gate_apply_kernel<<<(NN * 128 + 255) / 256, 256, 0, st>>>(x, h2, msum, NN);
}

// ---------------------------------------------------------------------------
extern "C" void kernel_launch(void* const* d_in, const int* in_sizes, int n_in,
                              void* d_out, int out_size, void* d_ws, size_t ws_size,
                              hipStream_t stream)
{
    (void)in_sizes; (void)n_in; (void)out_size; (void)ws_size;
    hipStream_t st = stream;

    const float* atom_pos    = (const float*)d_in[0];
    const float* atom_in     = (const float*)d_in[1];
    const float* bond_scalar = (const float*)d_in[2];
    int ii = 3;
    auto F = [&]() { return (const float*)d_in[ii++]; };
    // params leaves in jax tree order (dict keys sorted at each level)
    const float *aeW1=F(),*aeW2=F(),*aeW3=F(),*ae_b=F(),*aeb1=F(),*aeb2=F(),*aeb3=F(),*ae_g=F();
    const float *beW1=F(),*beW2=F(),*beW3=F(),*be_b=F(),*beb1=F(),*beb2=F(),*beb3=F(),*be_g=F();
    const float *blWa1=F(),*blWa2=F(),*blWf1=F(),*blWf2=F(),*blWg=F(),*blWo=F(),*blWv=F(),
                *blba1=F(),*blba2=F(),*blbf1=F(),*blbf2=F(),*blbg=F();
    const float *euW1=F(),*euW2=F(),*eu_b=F(),*eub1=F(),*eub2=F(),*eu_g=F();
    const float *puW=F(),*pu_b=F();
    const float *seW1=F(),*seW2=F(),*seW3=F(),*se_b=F(),*seb1=F(),*seb2=F(),*seb3=F(),*se_g=F();
    const float *slWa1=F(),*slWa2=F(),*slWf1=F(),*slWf2=F(),*slWg=F(),*slWo=F(),*slWv=F(),
                *slba1=F(),*slba2=F(),*slbf1=F(),*slbf2=F(),*slbg=F();
    const int* bond_ei = (const int*)d_in[ii++];
    const int* batch   = (const int*)d_in[ii++];
    const int* bsrc = bond_ei;
    const int* bdst = bond_ei + EBN;

    // workspace carve-up (floats)
    float* WS     = (float*)d_ws;
    float* x      = WS;                       // 8192*4*128
    float* vnode  = x      + (size_t)NN * 512;
    float* msum   = vnode  + (size_t)NN * 512;
    float* efb    = msum   + (size_t)NN * 512;   // 16128*128
    float* efs    = efb    + (size_t)EBN * 128;  // 81920*128
    float* h1     = efs    + (size_t)ESN * 128;  // 81920*128
    float* h2     = h1     + (size_t)ESN * 128;  // 81920*128
    float* rbfb   = h2     + (size_t)ESN * 128;  // 81920*32
    float* logits = rbfb   + (size_t)ESN * 32;   // 81920*4
    int*   nmax   = (int*)(logits + (size_t)ESN * 4);  // 8192*4
    float* nden   = (float*)(nmax + NN * 4);           // 8192*4
    int*   ssrc   = (int*)(nden + NN * 4);             // 81920
    int*   sdst   = ssrc + ESN;                        // 81920
    float* pos    = (float*)d_out;

    // ---- init: pos = atom_pos, x = 0, sdst = e / K ----
    copy_f32<<<(NN * 3 + 255) / 256, 256, 0, st>>>(pos, atom_pos, NN * 3);
    zero_f32<<<(NN * 512 + 255) / 256, 256, 0, st>>>(x, NN * 512);
    iota_div<<<(ESN + 255) / 256, 256, 0, st>>>(sdst, ESN, KNB);

    // ---- x0 = mlp3(atom_in, ae) -> x[:,0,:] ----
    gemm_plain(st, atom_in, 43, aeW1, aeb1, h1, 128, NN, 128, 43, 1, 0);
    gemm_plain(st, h1, 128, aeW2, aeb2, h2, 128, NN, 128, 128, 1, 0);
    launch_gemm(st, h2, nullptr, 128, 128, nullptr, nullptr, 0, 0, nullptr, nullptr, 0, 0,
                aeW3, aeb3, x, 512, NN, 128, 128, 0, 0);
    ln128_kernel<<<NN, 128, 0, st>>>(x, nullptr, ae_g, ae_b, 512);

    // ---- ef_b = mlp3([bond_scalar | rbf(blen,0,3,32)], be) ----
    bond_rbf_kernel<<<(EBN + 255) / 256, 256, 0, st>>>(atom_pos, bsrc, bdst, rbfb, EBN);
    launch_gemm(st, bond_scalar, nullptr, 3, 3, rbfb, nullptr, 32, 32,
                nullptr, nullptr, 0, 0, beW1, beb1, h1, 128, EBN, 128, 35, 1, 0);
    gemm_plain(st, h1, 128, beW2, beb2, h2, 128, EBN, 128, 128, 1, 0);
    gemm_plain(st, h2, 128, beW3, beb3, efb, 128, EBN, 128, 128, 0, 0);
    ln128_kernel<<<EBN, 128, 0, st>>>(efb, nullptr, be_g, be_b, 128);

    // ---- layers ----
    for (int l = 0; l < 4; ++l) {
        const size_t oW  = (size_t)l * 16384;   // 128x128
        const size_t oA1 = (size_t)l * 49152;   // 384x128
        const size_t ob  = (size_t)l * 128;
        AttnP bp = { blWa1 + oA1, blba1 + ob, blWa2 + (size_t)l * 512, blba2 + (size_t)l * 4,
                     blWv + oW, blWo + oW, blWf1 + oW, blbf1 + ob,
                     blWf2 + oW, blbf2 + ob, blWg + oW, blbg + ob };
        run_attn(st, x, efb, EBN, bsrc, bdst, bp, h1, h2, logits, nmax, nden, vnode, msum);

        // edge update: ef_b = LN(ef_b + relu([x0s|x0d|ef_b]@W1+b1)@W2+b2)
        launch_gemm(st, x, bsrc, 128, 512, x, bdst, 128, 512, efb, nullptr, 128, 128,
                    euW1 + oA1, eub1 + ob, h1, 128, EBN, 128, 384, 1, 0);
        gemm_plain(st, h1, 128, euW2 + oW, eub2 + ob, h2, 128, EBN, 128, 128, 0, 0);
        ln128_kernel<<<EBN, 128, 0, st>>>(efb, h2, eu_g + ob, eu_b + ob, 128);

        // KNN on current pos + slen rbf, then ef_s = mlp3(rbf, se[l])
        knn_kernel<<<(NN + 255) / 256, 256, 0, st>>>(pos, batch, ssrc, rbfb, NN);
        gemm_plain(st, rbfb, 32, seW1 + (size_t)l * 4096, seb1 + ob, h1, 128, ESN, 128, 32, 1, 0);
        gemm_plain(st, h1, 128, seW2 + oW, seb2 + ob, h2, 128, ESN, 128, 128, 1, 0);
        gemm_plain(st, h2, 128, seW3 + oW, seb3 + ob, efs, 128, ESN, 128, 128, 0, 0);
        ln128_kernel<<<ESN, 128, 0, st>>>(efs, nullptr, se_g + ob, se_b + ob, 128);

        AttnP sp = { slWa1 + oA1, slba1 + ob, slWa2 + (size_t)l * 512, slba2 + (size_t)l * 4,
                     slWv + oW, slWo + oW, slWf1 + oW, slbf1 + ob,
                     slWf2 + oW, slbf2 + ob, slWg + oW, slbg + ob };
        run_attn(st, x, efs, ESN, ssrc, sdst, sp, h1, h2, logits, nmax, nden, vnode, msum);

        // pos update: g = x0 @ puW + pub ; pos += sum_c x[:,1:4,c] * g[:,c]
        launch_gemm(st, x, nullptr, 128, 512, nullptr, nullptr, 0, 0, nullptr, nullptr, 0, 0,
                    puW + oW, pu_b + ob, h1, 128, NN, 128, 128, 0, 0);
        pos_update_kernel<<<NN, 32, 0, st>>>(pos, x, h1, NN);
    }
}
